// DualTranDec_10179072491957
// MI455X (gfx1250) — matrix operand order repaired and checked by reading, not compiled, not run on hardware
//
#include <hip/hip_runtime.h>
#include <cstdint>
#include <cstddef>

// ---------------------------------------------------------------------------
// Types for CDNA5 WMMA (wave32, 16x16x32 bf16 -> f32) + TDM descriptors
// ---------------------------------------------------------------------------
typedef __attribute__((ext_vector_type(8)))  __bf16 v8bf;
typedef __attribute__((ext_vector_type(16))) __bf16 v16bf;
typedef __attribute__((ext_vector_type(8)))  float  v8f;
typedef __attribute__((ext_vector_type(4)))  unsigned int u32x4;
typedef __attribute__((ext_vector_type(4)))  int          i32x4;
typedef __attribute__((ext_vector_type(8)))  int          i32x8;

constexpr int BSZ = 2, TQ = 1024, TK = 1024, DM = 512, HN = 8, DHD = 64, FFD = 2048;
constexpr int MROWS = BSZ * TQ;        // 2048 rows for every GEMM
constexpr float NEG_BIG = -1e30f;

// TDM weight staging: k-chunk of 128 bf16 per row, 64 rows, padded rows.
// 128 bf16 = 64 DWORDs -> pad_interval code 5 (64 DWORDs), pad 4 DWORDs (16B):
// row stride = 256 + 16 = 272 B  => bank rotation of 4 banks/row (conflict-free
// for 16 lanes reading consecutive rows with b128 accesses).
constexpr int KCH = 128;
constexpr int WROWB = 272;

static __device__ __forceinline__ unsigned short f2bf(float f) {
  unsigned u = __float_as_uint(f);
  unsigned r = (u + 0x7FFFu + ((u >> 16) & 1u)) >> 16;
  return (unsigned short)r;
}

static __device__ __forceinline__ v16bf join8(v8bf a, v8bf b) {
  return __builtin_shufflevector(a, b, 0,1,2,3,4,5,6,7,8,9,10,11,12,13,14,15);
}

// A-fragment: lanes 0-15 hold K {0..15}, lanes 16-31 hold K {16..31}.
// rowptr points at row*ldk + hi*16 + k0.
static __device__ __forceinline__ v16bf load_a_frag(const unsigned short* rowptr) {
  v8bf a = *(const v8bf*)(const void*)rowptr;
  v8bf b = *(const v8bf*)(const void*)(rowptr + 8);
  return join8(a, b);
}

// B-fragment: 16 contiguous bf16 (two 16B loads; works for global and LDS).
static __device__ __forceinline__ v16bf load_b_frag(const unsigned short* p) {
  v8bf a = *(const v8bf*)(const void*)p;
  v8bf b = *(const v8bf*)(const void*)(p + 8);
  return join8(a, b);
}

static __device__ __forceinline__ v8f wmma_bf16(v16bf a, v16bf b, v8f c) {
  return __builtin_amdgcn_wmma_f32_16x16x32_bf16(false, a, false, b, (short)0, c,
                                                 false, false);
}

// ---------------------------------------------------------------------------
// Tensor Data Mover: DMA a [64 rows x KCH cols] bf16 tile of W (row-major,
// leading dim K elements) into LDS at lds_off, padded 16B per 256B row.
// Issued once per wave that executes it (EXEC ignored); tracked by TENSORcnt.
// ---------------------------------------------------------------------------
static __device__ __forceinline__ void tdm_load_w(const unsigned short* gsrc,
                                                  unsigned lds_off, int K) {
  unsigned long long ga = (unsigned long long)gsrc;
  u32x4 g0;
  g0[0] = 1u;                                    // count=1 (valid user D#)
  g0[1] = lds_off;                               // LDS byte address
  g0[2] = (unsigned)(ga & 0xFFFFFFFFu);          // global_addr[31:0]
  g0[3] = (unsigned)((ga >> 32) & 0x01FFFFFFu)   // global_addr[56:32]
        | (2u << 30);                            // type = 2 ("image")
  unsigned td0 = (unsigned)K;                    // tensor_dim0 (elems/row)
  unsigned st0 = (unsigned)K;                    // tensor_dim0_stride
  unsigned td1 = 1u << 20;                       // tensor_dim1 (ample)
  i32x8 g1;
  g1[0] = (int)((1u << 16)                       // data_size = 1 -> 2 bytes
        |       (1u << 20)                       // pad_enable
        |       (5u << 22)                       // pad_interval: 64 DWORDs
        |       (3u << 25));                     // pad_amount: 4 DWORDs (16B)
  g1[1] = (int)((td0 & 0xFFFFu) << 16);          // tensor_dim0[15:0] @ bit48
  g1[2] = (int)(((td0 >> 16) & 0xFFFFu)          // tensor_dim0[31:16]
        |       ((td1 & 0xFFFFu) << 16));        // tensor_dim1[15:0] @ bit80
  g1[3] = (int)(((td1 >> 16) & 0xFFFFu)          // tensor_dim1[31:16]
        |       ((unsigned)KCH << 16));          // tile_dim0 = 128 @ bit112
  g1[4] = (int)64;                               // tile_dim1 = 64 rows
  g1[5] = (int)st0;                              // tensor_dim0_stride[31:0]
  g1[6] = 0;                                     // stride hi / dim1_stride lo
  g1[7] = 0;
  i32x4 gz; gz[0] = 0; gz[1] = 0; gz[2] = 0; gz[3] = 0;
#if defined(__clang_major__) && (__clang_major__ >= 23)
  i32x8 gz8; gz8[0]=0; gz8[1]=0; gz8[2]=0; gz8[3]=0; gz8[4]=0; gz8[5]=0; gz8[6]=0; gz8[7]=0;
  __builtin_amdgcn_tensor_load_to_lds(g0, g1, gz, gz, gz8, 0);
#else
  __builtin_amdgcn_tensor_load_to_lds(g0, g1, gz, gz, 0);
#endif
}

// ---------------------------------------------------------------------------
// fp32 -> bf16 conversion (4 elems/thread), and x init (copy + bf16 mirror)
// ---------------------------------------------------------------------------
__global__ void k_cvt_bf16(const float* __restrict__ s, unsigned short* __restrict__ d,
                           int n) {
  int i = (blockIdx.x * 256 + threadIdx.x) * 4;
  if (i < n) {
    float4 v = *(const float4*)(s + i);
    unsigned a = (unsigned)f2bf(v.x) | ((unsigned)f2bf(v.y) << 16);
    unsigned b = (unsigned)f2bf(v.z) | ((unsigned)f2bf(v.w) << 16);
    *(uint2*)(void*)(d + i) = make_uint2(a, b);
  }
}

__global__ void k_init_x(const float* __restrict__ q, float* __restrict__ x,
                         unsigned short* __restrict__ xb, int n) {
  int i = (blockIdx.x * 256 + threadIdx.x) * 4;
  if (i < n) {
    float4 v = *(const float4*)(q + i);
    *(float4*)(x + i) = v;
    unsigned a = (unsigned)f2bf(v.x) | ((unsigned)f2bf(v.y) << 16);
    unsigned b = (unsigned)f2bf(v.z) | ((unsigned)f2bf(v.w) << 16);
    *(uint2*)(void*)(xb + i) = make_uint2(a, b);
  }
}

// ---------------------------------------------------------------------------
// GEMM: C[m,n] = sum_k A[m,k] * W[n,k]  (+bias, *scale, epilogue variant)
// A: [M,K] bf16 row-major (streamed from global), W: [N,K] bf16 row-major,
// staged through LDS by the Tensor Data Mover, double-buffered in KCH chunks.
// Block 256 = 8 waves, wave tile 16x64, block tile 128x64. Grid (N/64, M/128).
// B-fragments are hoisted so the 8 ds_load_b128 issue as one clause with
// incremental dscnt waits, overlapping LDS latency with the WMMA chain.
// EPI: 0 = f32 [M,N]; 1 = bf16 [M,N] + ReLU; 2 = bf16 [B,H,T,DH];
//      3 = bf16 [B,H,DH,T] (transposed, packed b128 stores)
// ---------------------------------------------------------------------------
template <int EPI>
__global__ void __launch_bounds__(256)
k_gemm(const unsigned short* __restrict__ A, const unsigned short* __restrict__ W,
       const float* __restrict__ bias, void* __restrict__ out,
       int N, int K, float scale) {
  __shared__ __align__(16) unsigned char wsm[2][64 * WROWB];

  const int lane = threadIdx.x & 31;
  const int wv   = threadIdx.x >> 5;
  const int l    = lane & 15;
  const int hi   = lane >> 4;
  const int m0   = blockIdx.y * 128 + wv * 16;
  const int n0   = blockIdx.x * 64;

  v8f acc[4];
#pragma unroll
  for (int j = 0; j < 4; ++j)
#pragma unroll
    for (int r = 0; r < 8; ++r) acc[j][r] = 0.f;

  const unsigned short* arow = A + (size_t)(m0 + l) * K + hi * 16;
  const unsigned short* wpan = W + (size_t)n0 * K;
  const unsigned lds0 = (unsigned)(unsigned long long)(void*)&wsm[0][0];
  const unsigned lds1 = (unsigned)(unsigned long long)(void*)&wsm[1][0];

  const int nch = K / KCH;
  if (wv == 0) tdm_load_w(wpan, lds0, K);        // prime chunk 0

  int buf = 0;
  for (int c = 0; c < nch; ++c) {
    if (wv == 0) __builtin_amdgcn_s_wait_tensorcnt(0);
    __syncthreads();                             // chunk c visible to all waves
    if (wv == 0 && (c + 1) < nch)                // prefetch chunk c+1 via TDM
      tdm_load_w(wpan + (size_t)(c + 1) * KCH, buf ? lds0 : lds1, K);

    const unsigned char* wb = wsm[buf];
#pragma unroll
    for (int kk = 0; kk < KCH; kk += 32) {
      v16bf a = load_a_frag(arow + c * KCH + kk);
      v16bf bfr[4];
#pragma unroll
      for (int j = 0; j < 4; ++j)                // issue all LDS loads first
        bfr[j] = load_b_frag((const unsigned short*)(const void*)
            (wb + (size_t)(16 * j + l) * WROWB + (size_t)(kk + hi * 16) * 2));
#pragma unroll
      for (int j = 0; j < 4; ++j)                // then the WMMA chain
        acc[j] = wmma_bf16(a, bfr[j], acc[j]);
    }
    buf ^= 1;
  }

#pragma unroll
  for (int j = 0; j < 4; ++j) {
    const int n = n0 + 16 * j + l;
    const float bn = bias[n];
    float vals[8];
#pragma unroll
    for (int r = 0; r < 8; ++r) vals[r] = (acc[j][r] + bn) * scale;

    if (EPI == 0) {
      float* o = (float*)out;
#pragma unroll
      for (int r = 0; r < 8; ++r)
        o[(size_t)(m0 + r + 8 * hi) * N + n] = vals[r];
    } else if (EPI == 1) {
      unsigned short* o = (unsigned short*)out;
#pragma unroll
      for (int r = 0; r < 8; ++r)
        o[(size_t)(m0 + r + 8 * hi) * N + n] = f2bf(fmaxf(vals[r], 0.f));
    } else if (EPI == 2) {
      const int b = m0 / TQ, t0 = m0 % TQ;
      const int h = n >> 6, dh = n & 63;
      unsigned short* o = (unsigned short*)out +
          (((size_t)(b * HN + h) * TQ) + t0) * DHD + dh;
#pragma unroll
      for (int r = 0; r < 8; ++r)
        o[(size_t)(r + 8 * hi) * DHD] = f2bf(vals[r]);
    } else {  // EPI == 3 : per-head transposed V, contiguous along t -> b128
      const int b = m0 / TQ, t0 = m0 % TQ;
      const int h = n >> 6, dh = n & 63;
      unsigned pk[4];
#pragma unroll
      for (int r2 = 0; r2 < 4; ++r2)
        pk[r2] = (unsigned)f2bf(vals[2 * r2]) |
                 ((unsigned)f2bf(vals[2 * r2 + 1]) << 16);
      unsigned short* o = (unsigned short*)out +
          ((size_t)(b * HN + h) * DHD + dh) * TK + t0 + 8 * hi;
      *(uint4*)(void*)o = make_uint4(pk[0], pk[1], pk[2], pk[3]);
    }
  }
}

// ---------------------------------------------------------------------------
// Fused flash attention (templated on CAUSAL to keep masking branchless).
// One wave = one (b, h, 16-query tile). Q: [B,H,TQ,DH] bf16 (pre-scaled 1/8),
// K: [B,H,TK,DH] bf16, Vt: [B,H,DH,TK] bf16. Out: attn_out [B*TQ, HID] bf16.
// 32 keys/iter: 4 score WMMAs, online softmax (16-lane butterflies over the
// C-layout lane halves), P re-fragmented through per-wave LDS, 4 PV WMMAs.
// ---------------------------------------------------------------------------
template <int CAUSAL>
__global__ void __launch_bounds__(128)
k_flash(const unsigned short* __restrict__ Q, const unsigned short* __restrict__ Kh,
        const unsigned short* __restrict__ Vt, const float* __restrict__ kmask,
        unsigned short* __restrict__ Oout) {
  __shared__ __align__(32) unsigned short Plds[4][16 * 32];

  const int lane = threadIdx.x & 31;
  const int wv   = threadIdx.x >> 5;
  const int l    = lane & 15;
  const int hi   = lane >> 4;
  const int w    = blockIdx.x * 4 + wv;
  const int nqt  = TQ / 16;
  const int qt   = w % nqt;
  const int h    = (w / nqt) % HN;
  const int b    = w / (nqt * HN);
  const int q0   = qt * 16;

  const unsigned short* Qp = Q + ((size_t)(b * HN + h) * TQ + q0 + l) * DHD + hi * 16;
  const v16bf aq0 = load_a_frag(Qp);        // dh 0..31
  const v16bf aq1 = load_a_frag(Qp + 32);   // dh 32..63
  const unsigned short* Kp = Kh + (size_t)(b * HN + h) * TK * DHD;
  const unsigned short* Vp = Vt + (size_t)(b * HN + h) * DHD * TK;
  const float* km = kmask + (size_t)b * TK;

  float mi[8], li[8];
  v8f acc[4];
#pragma unroll
  for (int r = 0; r < 8; ++r) { mi[r] = NEG_BIG; li[r] = 0.f; }
#pragma unroll
  for (int d = 0; d < 4; ++d)
#pragma unroll
    for (int r = 0; r < 8; ++r) acc[d][r] = 0.f;

  const int ktend = CAUSAL ? (q0 + 16) : TK;
  for (int kt = 0; kt < ktend; kt += 32) {
    // ---- issue all K-tile loads first (one clause), then the WMMA chains ----
    v16bf kb[2][2];
#pragma unroll
    for (int j = 0; j < 2; ++j) {
      const unsigned short* kc = Kp + (size_t)(kt + 16 * j + l) * DHD + hi * 16;
      kb[j][0] = load_b_frag(kc);          // dh chunk 0..31
      kb[j][1] = load_b_frag(kc + 32);     // dh chunk 32..63
    }
    v8f s[2];
#pragma unroll
    for (int j = 0; j < 2; ++j) {
      v8f z;
#pragma unroll
      for (int r = 0; r < 8; ++r) z[r] = 0.f;
      s[j] = wmma_bf16(aq0, kb[j][0], z);
      s[j] = wmma_bf16(aq1, kb[j][1], s[j]);
      const bool kmz = (km[kt + 16 * j + l] == 0.f);  // branchless mask
#pragma unroll
      for (int r = 0; r < 8; ++r) {
        bool bad = kmz;
        if (CAUSAL) bad = bad | ((kt + 16 * j + l) > (q0 + r + 8 * hi));
        s[j][r] = bad ? NEG_BIG : s[j][r];
      }
    }
    // ---- online softmax over the 32-key block (rows live in lane halves) ----
    float bmax[8];
#pragma unroll
    for (int r = 0; r < 8; ++r) bmax[r] = fmaxf(s[0][r], s[1][r]);
    for (int msk = 1; msk < 16; msk <<= 1)
#pragma unroll
      for (int r = 0; r < 8; ++r)
        bmax[r] = fmaxf(bmax[r], __shfl_xor(bmax[r], msk, 32));
    float alpha[8];
#pragma unroll
    for (int r = 0; r < 8; ++r) {
      float mn = fmaxf(mi[r], bmax[r]);
      alpha[r] = __expf(mi[r] - mn);
      mi[r] = mn;
    }
#pragma unroll
    for (int j = 0; j < 2; ++j)
#pragma unroll
      for (int r = 0; r < 8; ++r) s[j][r] = __expf(s[j][r] - mi[r]);
    float rs[8];
#pragma unroll
    for (int r = 0; r < 8; ++r) rs[r] = s[0][r] + s[1][r];
    for (int msk = 1; msk < 16; msk <<= 1)
#pragma unroll
      for (int r = 0; r < 8; ++r) rs[r] += __shfl_xor(rs[r], msk, 32);
#pragma unroll
    for (int r = 0; r < 8; ++r) li[r] = li[r] * alpha[r] + rs[r];
#pragma unroll
    for (int d = 0; d < 4; ++d)
#pragma unroll
      for (int r = 0; r < 8; ++r) acc[d][r] *= alpha[r];

    // ---- bounce P through LDS: C-layout -> A-fragment (row-major 16x32) ----
#pragma unroll
    for (int j = 0; j < 2; ++j)
#pragma unroll
      for (int r = 0; r < 8; ++r)
        Plds[wv][(r + 8 * hi) * 32 + 16 * j + l] = f2bf(s[j][r]);
    v16bf ap = load_a_frag(&Plds[wv][l * 32 + hi * 16]);

    v16bf bv[4];                               // hoist V fragments (one clause)
#pragma unroll
    for (int d = 0; d < 4; ++d)
      bv[d] = load_b_frag(Vp + (size_t)(16 * d + l) * TK + kt + hi * 16);
#pragma unroll
    for (int d = 0; d < 4; ++d)
      acc[d] = wmma_bf16(ap, bv[d], acc[d]);
  }

  float inv[8];
#pragma unroll
  for (int r = 0; r < 8; ++r) inv[r] = 1.f / li[r];
#pragma unroll
  for (int d = 0; d < 4; ++d)
#pragma unroll
    for (int r = 0; r < 8; ++r) {
      size_t idx = (size_t)(b * TQ + q0 + r + 8 * hi) * (HN * DHD) +
                   h * DHD + 16 * d + l;
      Oout[idx] = f2bf(acc[d][r] * inv[r]);
    }
}

// ---------------------------------------------------------------------------
// Residual + mask + LayerNorm. One block (256 thr) per row of 512.
// x_new = LN(x + y*mask) * g + b; writes f32 x (in place) and bf16 mirror.
// ---------------------------------------------------------------------------
__global__ void __launch_bounds__(256)
k_res_ln(float* __restrict__ x, const float* __restrict__ y,
         const float* __restrict__ mask, const float* __restrict__ g,
         const float* __restrict__ be, unsigned short* __restrict__ xb) {
  const int row = blockIdx.x;
  const int tid = threadIdx.x;
  const float mk = mask[row];
  float* xr = x + (size_t)row * DM;
  const float* yr = y + (size_t)row * DM;

  float v0 = xr[tid] + yr[tid] * mk;
  float v1 = xr[tid + 256] + yr[tid + 256] * mk;
  float s = v0 + v1, s2 = v0 * v0 + v1 * v1;
  for (int msk = 16; msk >= 1; msk >>= 1) {
    s  += __shfl_xor(s, msk, 32);
    s2 += __shfl_xor(s2, msk, 32);
  }
  __shared__ float red[16];
  const int wv = tid >> 5, lane = tid & 31;
  if (lane == 0) { red[wv] = s; red[8 + wv] = s2; }
  __syncthreads();
  if (tid == 0) {
    float ts = 0.f, ts2 = 0.f;
    for (int i = 0; i < 8; ++i) { ts += red[i]; ts2 += red[8 + i]; }
    red[0] = ts; red[1] = ts2;
  }
  __syncthreads();
  const float mean = red[0] * (1.f / DM);
  const float var  = red[1] * (1.f / DM) - mean * mean;
  const float rinv = rsqrtf(var + 1e-5f);
  const float o0 = (v0 - mean) * rinv * g[tid] + be[tid];
  const float o1 = (v1 - mean) * rinv * g[tid + 256] + be[tid + 256];
  xr[tid] = o0;
  xr[tid + 256] = o1;
  unsigned short* xbr = xb + (size_t)row * DM;
  xbr[tid] = f2bf(o0);
  xbr[tid + 256] = f2bf(o1);
}

// ---------------------------------------------------------------------------
// Host orchestration
// ---------------------------------------------------------------------------
extern "C" void kernel_launch(void* const* d_in, const int* in_sizes, int n_in,
                              void* d_out, int out_size, void* d_ws, size_t ws_size,
                              hipStream_t stream) {
  (void)in_sizes; (void)n_in; (void)out_size; (void)ws_size;

  const float* query  = (const float*)d_in[0];
  const float* key1   = (const float*)d_in[1];
  const float* key2   = (const float*)d_in[2];
  const float* qmask  = (const float*)d_in[3];
  const float* kmask1 = (const float*)d_in[4];
  const float* kmask2 = (const float*)d_in[5];

  char* ws = (char*)d_ws;
  size_t off = 0;
  auto alloc = [&](size_t bytes) -> void* {
    void* p = ws + off;
    off = (off + bytes + 255) & ~(size_t)255;
    return p;
  };

  unsigned short* xb   = (unsigned short*)alloc((size_t)MROWS * DM * 2);
  unsigned short* kb1  = (unsigned short*)alloc((size_t)MROWS * DM * 2);
  unsigned short* kb2  = (unsigned short*)alloc((size_t)MROWS * DM * 2);
  unsigned short* Qh   = (unsigned short*)alloc((size_t)MROWS * DM * 2);
  unsigned short* Kh   = (unsigned short*)alloc((size_t)MROWS * DM * 2);
  unsigned short* Vt   = (unsigned short*)alloc((size_t)MROWS * DM * 2);
  unsigned short* aout = (unsigned short*)alloc((size_t)MROWS * DM * 2);
  float*          ybuf = (float*)alloc((size_t)MROWS * DM * 4);
  unsigned short* ffh  = (unsigned short*)alloc((size_t)MROWS * FFD * 2);
  float* x = (float*)d_out;  // f32 activations live in the output buffer

  auto cvtW = [&](const void* src, size_t n) -> const unsigned short* {
    unsigned short* dst = (unsigned short*)alloc(n * 2);
    k_cvt_bf16<<<dim3((unsigned)(n / 1024)), dim3(256), 0, stream>>>(
        (const float*)src, dst, (int)n);
    return dst;
  };

  struct AttnW {
    const unsigned short *wq, *wk, *wv, *wo;
    const float *bq, *bk, *bv, *bo, *g, *be;
  };
  struct FFW {
    const unsigned short *w1, *w2;
    const float *b1, *b2, *g, *be;
  };
  AttnW attn[4][3];
  FFW ffw[4];
  for (int L = 0; L < 4; ++L) {
    const int base = 6 + L * 36;
    for (int a = 0; a < 3; ++a) {
      const int p = base + a * 10;
      attn[L][a].wq = cvtW(d_in[p + 0], (size_t)DM * DM);
      attn[L][a].bq = (const float*)d_in[p + 1];
      attn[L][a].wk = cvtW(d_in[p + 2], (size_t)DM * DM);
      attn[L][a].bk = (const float*)d_in[p + 3];
      attn[L][a].wv = cvtW(d_in[p + 4], (size_t)DM * DM);
      attn[L][a].bv = (const float*)d_in[p + 5];
      attn[L][a].wo = cvtW(d_in[p + 6], (size_t)DM * DM);
      attn[L][a].bo = (const float*)d_in[p + 7];
      attn[L][a].g  = (const float*)d_in[p + 8];
      attn[L][a].be = (const float*)d_in[p + 9];
    }
    const int p = base + 30;
    ffw[L].w1 = cvtW(d_in[p + 0], (size_t)FFD * DM);
    ffw[L].b1 = (const float*)d_in[p + 1];
    ffw[L].w2 = cvtW(d_in[p + 2], (size_t)DM * FFD);
    ffw[L].b2 = (const float*)d_in[p + 3];
    ffw[L].g  = (const float*)d_in[p + 4];
    ffw[L].be = (const float*)d_in[p + 5];
  }

  const int nelem = MROWS * DM;
  k_cvt_bf16<<<dim3(nelem / 1024), dim3(256), 0, stream>>>(key1, kb1, nelem);
  k_cvt_bf16<<<dim3(nelem / 1024), dim3(256), 0, stream>>>(key2, kb2, nelem);
  k_init_x<<<dim3(nelem / 1024), dim3(256), 0, stream>>>(query, x, xb, nelem);

  auto gemm = [&](int epi, int N, int K, const unsigned short* A,
                  const unsigned short* W, const float* bias, void* out,
                  float scale) {
    dim3 g(N / 64, MROWS / 128);
    switch (epi) {
      case 0: k_gemm<0><<<g, 256, 0, stream>>>(A, W, bias, out, N, K, scale); break;
      case 1: k_gemm<1><<<g, 256, 0, stream>>>(A, W, bias, out, N, K, scale); break;
      case 2: k_gemm<2><<<g, 256, 0, stream>>>(A, W, bias, out, N, K, scale); break;
      default: k_gemm<3><<<g, 256, 0, stream>>>(A, W, bias, out, N, K, scale); break;
    }
  };

  auto mha = [&](const unsigned short* qsrc, const unsigned short* kvsrc,
                 const AttnW& a, const float* kmsk, int causal) {
    gemm(2, DM, DM, qsrc,  a.wq, a.bq, Qh, 0.125f);  // Q, pre-scaled DH^-0.5
    gemm(2, DM, DM, kvsrc, a.wk, a.bk, Kh, 1.0f);    // K per-head row-major
    gemm(3, DM, DM, kvsrc, a.wv, a.bv, Vt, 1.0f);    // V per-head transposed
    dim3 fg(BSZ * HN * (TQ / 16) / 4);
    if (causal)
      k_flash<1><<<fg, 128, 0, stream>>>(Qh, Kh, Vt, kmsk, aout);
    else
      k_flash<0><<<fg, 128, 0, stream>>>(Qh, Kh, Vt, kmsk, aout);
    gemm(0, DM, DM, aout, a.wo, a.bo, ybuf, 1.0f);   // output projection (f32)
    k_res_ln<<<dim3(MROWS), dim3(256), 0, stream>>>(x, ybuf, qmask, a.g, a.be, xb);
  };

  for (int L = 0; L < 4; ++L) {
    mha(xb, xb,  attn[L][0], qmask,  1);   // causal self-attention
    mha(xb, kb1, attn[L][1], kmask1, 0);   // cross-attention 1
    mha(xb, kb2, attn[L][2], kmask2, 0);   // cross-attention 2
    gemm(1, FFD, DM, xb,  ffw[L].w1, ffw[L].b1, ffh,  1.0f);  // FF1 + ReLU
    gemm(0, DM, FFD, ffh, ffw[L].w2, ffw[L].b2, ybuf, 1.0f);  // FF2
    k_res_ln<<<dim3(MROWS), dim3(256), 0, stream>>>(x, ybuf, qmask,
                                                    ffw[L].g, ffw[L].be, xb);
  }
}
